// VectorQuantizer_81355270521218
// MI455X (gfx1250) — matrix operand order repaired and checked by reading, not compiled, not run on hardware
//
#include <hip/hip_runtime.h>
#include <hip/hip_bf16.h>

// CDNA5 / gfx1250 VQ quantizer: bf16 split-precision WMMA distance GEMM.
typedef __attribute__((ext_vector_type(16))) __bf16 v16bf;
typedef __attribute__((ext_vector_type(8)))  __bf16 v8bf;
typedef __attribute__((ext_vector_type(8)))  float  v8f;

#define NUM_CODES 1024
#define DIM       64
#define NROWS     65536      // 64 * 32 * 32
#define QELEMS    4194304    // 64 * 64 * 32 * 32

static __device__ __forceinline__ unsigned short f32_to_bf16_rtne(float f) {
  unsigned u = __float_as_uint(f);
  u += 0x7FFFu + ((u >> 16) & 1u);
  return (unsigned short)(u >> 16);
}
static __device__ __forceinline__ float bf16bits_to_f32(unsigned short h) {
  return __uint_as_float(((unsigned)h) << 16);
}
static __device__ __forceinline__ __bf16 bfbits(unsigned short u) {
  __bf16 b; __builtin_memcpy(&b, &u, 2); return b;
}

// ---------------------------------------------------------------------------
// Kernel 0a: split embedding scaled by -2 into bf16 hi/lo (one-time prep).
// eHi/eLo hold bf16(-2*e) so the WMMA chain accumulates -2*x.e directly.
__global__ void vq_split_embedding(const float* __restrict__ emb,
                                   unsigned short* __restrict__ eHi,
                                   unsigned short* __restrict__ eLo) {
  int i = blockIdx.x * 256 + threadIdx.x;            // 65536 elements
  float t = -2.0f * emb[i];
  unsigned short h = f32_to_bf16_rtne(t);
  eHi[i] = h;
  eLo[i] = f32_to_bf16_rtne(t - bf16bits_to_f32(h));
}

// Kernel 0b: per-code squared norms in f32.
__global__ void vq_enorm(const float* __restrict__ emb,
                         float* __restrict__ enorm) {
  int k = blockIdx.x * 256 + threadIdx.x;            // 1024 codes
  const float* e = emb + (size_t)k * DIM;
  float s = 0.f;
#pragma unroll
  for (int d = 0; d < DIM; ++d) { float v = e[d]; s += v * v; }
  enorm[k] = s;
}

// ---------------------------------------------------------------------------
// Kernel 1: main. Block = 256 threads = 8 waves; each wave owns a 16-row tile.
// Score tile C[16 codes x 16 rows] = ||e||^2 - 2 x.e via
//   c = wmma(aHi,bHi,c); c = wmma(aHi,bLo,c); c = wmma(aLo,bHi,c)  per 32-dim chunk.
__global__ void __launch_bounds__(256)
vq_main(const float* __restrict__ lat, const float* __restrict__ emb,
        const unsigned short* __restrict__ eHi,
        const unsigned short* __restrict__ eLo,
        const float* __restrict__ enormG,
        float* __restrict__ quant,     // d_out + 1, NCHW
        float* __restrict__ idxOut,    // d_out + 1 + QELEMS, idx as float
        float* __restrict__ partial) { // per-block loss partials (d_ws)
  __shared__ __align__(16) float sEnorm[NUM_CODES];
  __shared__ float sX[8][16][68];      // padded rows -> conflict-free
  __shared__ float sLoss[8];

  const int tid  = threadIdx.x;
  const int wave = tid >> 5;
  const int lane = tid & 31;

  for (int i = tid; i < NUM_CODES; i += 256) sEnorm[i] = enormG[i];

  // --- stage 16 rows x 64 dims of X (NCHW gather) into LDS, coalesced ---
  const int tile = blockIdx.x * 8 + wave;            // 0..4095
  const int n0 = tile * 16;
  const int b  = n0 >> 10;                           // image index
  const int s0 = n0 & 1023;                          // h*W+w base
  const float* src = lat + (size_t)b * 65536 + s0;
#pragma unroll
  for (int it = 0; it < 8; ++it) {
    int id = it * 32 + lane;                         // 0..255 -> (d, rowgroup)
    int d  = id >> 2;
    int rg = id & 3;
    float4 v = *(const float4*)(src + (size_t)d * 1024 + rg * 4);
    sX[wave][rg * 4 + 0][d] = v.x;
    sX[wave][rg * 4 + 1][d] = v.y;
    sX[wave][rg * 4 + 2][d] = v.z;
    sX[wave][rg * 4 + 3][d] = v.w;
  }
  __syncthreads();

  const int r    = lane & 15;                        // row within tile / code within tile
  const int off8 = (lane < 16) ? 0 : 8;              // A-layout K-group base
  const int g16  = (lane < 16) ? 0 : 16;             // unused (B is built with off8)

  // --- B operand: x row in bf16 hi/lo, both 32-dim chunks (converted once) ---
  v16bf bH[2], bL[2];
  const float* xrow = &sX[wave][r][0];
#pragma unroll
  for (int ch = 0; ch < 2; ++ch) {
#pragma unroll
    for (int j = 0; j < 16; ++j) {
      float f = xrow[32 * ch + off8 + (j < 8 ? j : j + 8)];
      unsigned short h = f32_to_bf16_rtne(f);
      bH[ch][j] = bfbits(h);
      bL[ch][j] = bfbits(f32_to_bf16_rtne(f - bf16bits_to_f32(h)));
    }
  }

  float best = 3.4e38f;
  int bestIdx = 0;

#pragma unroll 2
  for (int k0 = 0; k0 < NUM_CODES; k0 += 16) {
    // A operand: code (k0 + r), dims in ISA A-layout groups (16B loads, L2-hot)
    const unsigned short* pH = eHi + (size_t)(k0 + r) * DIM;
    const unsigned short* pL = eLo + (size_t)(k0 + r) * DIM;
    v16bf aH[2], aL[2];
#pragma unroll
    for (int ch = 0; ch < 2; ++ch) {
      v8bf h0 = *(const v8bf*)(const void*)(pH + 32 * ch + off8);
      v8bf h1 = *(const v8bf*)(const void*)(pH + 32 * ch + off8 + 16);
      v8bf l0 = *(const v8bf*)(const void*)(pL + 32 * ch + off8);
      v8bf l1 = *(const v8bf*)(const void*)(pL + 32 * ch + off8 + 16);
      aH[ch] = __builtin_shufflevector(h0, h1, 0,1,2,3,4,5,6,7,8,9,10,11,12,13,14,15);
      aL[ch] = __builtin_shufflevector(l0, l1, 0,1,2,3,4,5,6,7,8,9,10,11,12,13,14,15);
    }

    // init accumulator with ||e||^2 per the C-layout (c[v] -> code k0+off8+v)
    v8f c;
    const float4 e0 = *(const float4*)&sEnorm[k0 + off8];
    const float4 e1 = *(const float4*)&sEnorm[k0 + off8 + 4];
    c[0] = e0.x; c[1] = e0.y; c[2] = e0.z; c[3] = e0.w;
    c[4] = e1.x; c[5] = e1.y; c[6] = e1.z; c[7] = e1.w;

    c = __builtin_amdgcn_wmma_f32_16x16x32_bf16(false, aH[0], false, bH[0], (short)0, c, false, false);
    c = __builtin_amdgcn_wmma_f32_16x16x32_bf16(false, aH[0], false, bL[0], (short)0, c, false, false);
    c = __builtin_amdgcn_wmma_f32_16x16x32_bf16(false, aL[0], false, bH[0], (short)0, c, false, false);
    c = __builtin_amdgcn_wmma_f32_16x16x32_bf16(false, aH[1], false, bH[1], (short)0, c, false, false);
    c = __builtin_amdgcn_wmma_f32_16x16x32_bf16(false, aH[1], false, bL[1], (short)0, c, false, false);
    c = __builtin_amdgcn_wmma_f32_16x16x32_bf16(false, aL[1], false, bH[1], (short)0, c, false, false);

#pragma unroll
    for (int v = 0; v < 8; ++v) {        // branchless argmin, first-min tie rule
      float s = c[v];
      int ix = k0 + off8 + v;
      bool lt = s < best;
      best    = lt ? s  : best;
      bestIdx = lt ? ix : bestIdx;
    }
  }

  // merge the two M-halves (lane L vs L+16 hold same row, disjoint code sets)
  {
    float ob = __shfl_xor(best, 16, 32);
    int   oi = __shfl_xor(bestIdx, 16, 32);
    bool take = (ob < best) || ((ob == best) && (oi < bestIdx));
    best    = take ? ob : best;
    bestIdx = take ? oi : bestIdx;
  }
  const int rowIdx = __shfl(bestIdx, r, 32);   // broadcast row r's winner to both halves

  // --- outputs: quant (NCHW scatter, 2x64B coalesced per step) + exact f32 loss ---
  const int dh = lane >> 4;                    // even/odd dim half
  const float* er = emb + (size_t)rowIdx * DIM;
  float* qb = quant + (size_t)b * 65536 + s0 + r;
  float acc = 0.f;
#pragma unroll 4
  for (int dd = 0; dd < 32; ++dd) {
    int d = dd * 2 + dh;
    float ev = er[d];                          // L2-hot gather of winning code
    float xv = sX[wave][r][d];
    float df = xv - ev;
    acc += df * df;
    qb[(size_t)d * 1024] = ev;
  }
  if (lane < 16) idxOut[n0 + r] = (float)rowIdx;

#pragma unroll
  for (int m = 16; m >= 1; m >>= 1) acc += __shfl_xor(acc, m, 32);
  if (lane == 0) sLoss[wave] = acc;
  __syncthreads();
  if (tid == 0) {
    float s = 0.f;
#pragma unroll
    for (int w = 0; w < 8; ++w) s += sLoss[w];
    partial[blockIdx.x] = s;
  }
}

// ---------------------------------------------------------------------------
// Kernel 2: deterministic fixed-order reduction of 512 block partials.
__global__ void vq_reduce(const float* __restrict__ partial,
                          float* __restrict__ out) {
  __shared__ float s[256];
  int t = threadIdx.x;
  s[t] = partial[t] + partial[t + 256];
  __syncthreads();
  for (int m = 128; m >= 1; m >>= 1) {
    if (t < m) s[t] += s[t + m];
    __syncthreads();
  }
  if (t == 0) out[0] = 0.25f * s[0] / 4194304.0f;   // COMMITMENT_COST * mean
}

// ---------------------------------------------------------------------------
extern "C" void kernel_launch(void* const* d_in, const int* in_sizes, int n_in,
                              void* d_out, int out_size, void* d_ws, size_t ws_size,
                              hipStream_t stream) {
  const float* lat = (const float*)d_in[0];   // [64,64,32,32] f32
  const float* emb = (const float*)d_in[1];   // [1024,64] f32
  float* out = (float*)d_out;                 // [loss | quant(QELEMS) | idx(NROWS)]

  unsigned char* ws = (unsigned char*)d_ws;
  unsigned short* eHi = (unsigned short*)(ws);            // 131072 B
  unsigned short* eLo = (unsigned short*)(ws + 131072);   // 131072 B
  float* enorm   = (float*)(ws + 262144);                 //   4096 B
  float* partial = (float*)(ws + 266240);                 //   2048 B

  vq_split_embedding<<<256, 256, 0, stream>>>(emb, eHi, eLo);
  vq_enorm<<<4, 256, 0, stream>>>(emb, enorm);
  vq_main<<<512, 256, 0, stream>>>(lat, emb, eHi, eLo, enorm,
                                   out + 1, out + 1 + QELEMS, partial);
  vq_reduce<<<1, 256, 0, stream>>>(partial, out);
}
